// ImagePatchGraphConstructor_88192858456451
// MI455X (gfx1250) — compile-verified, matrix operand order checked by reading, not run on hardware
//
#include <hip/hip_runtime.h>
#include <hip/hip_bf16.h>

typedef __attribute__((ext_vector_type(16))) _Float16 v16h;
typedef __attribute__((ext_vector_type(8)))  float    v8f;

#define B_      16
#define C_      3
#define H_      512
#define W_      512
#define P_      16
#define S_      8
#define NHW     63            // (512-16)/8 + 1
#define NTOT    (NHW * NHW)   // 3969
#define KNN     8
#define KP1     (KNN + 1)     // 9 (top-k incl. self)
#define QTILES  249           // ceil(3969/16)
#define FEAT    (C_ * P_ * P_) // 768

// ---------------------------------------------------------------------------
// Kernel 1: im2col patch extraction (pure bandwidth, float4-aligned both ways)
// patches[b][n][c*256 + py*16 + px] = x[b][c][i*8+py][j*8+px], n = i*63 + j
// ---------------------------------------------------------------------------
__global__ __launch_bounds__(256) void patches_kernel(const float* __restrict__ x,
                                                      float* __restrict__ out) {
  unsigned t = blockIdx.x * 256u + threadIdx.x;   // exactly 12,192,768 threads
  int px4 = t & 3;          // which float4 within the 16-float patch row
  unsigned g = t >> 2;      // (b, n, c, py) group
  int py = g & 15;  g >>= 4;
  int c  = g % 3;   g /= 3;
  int n  = g % NTOT;
  int b  = g / NTOT;
  int i = n / NHW;
  int j = n % NHW;
  const float4 v = *reinterpret_cast<const float4*>(
      x + (((size_t)(b * C_ + c) * H_ + (i * S_ + py)) * W_ + j * S_ + px4 * 4));
  *reinterpret_cast<float4*>(
      out + ((size_t)(b * NTOT + n) * FEAT + c * (P_ * P_) + py * P_ + px4 * 4)) = v;
}

// ---------------------------------------------------------------------------
// Kernel 2: positions[b][n] = (i, j) — pure arithmetic
// ---------------------------------------------------------------------------
__global__ __launch_bounds__(256) void pos_kernel(float* __restrict__ out) {
  int t = blockIdx.x * 256 + threadIdx.x;
  if (t >= B_ * NTOT) return;
  int n = t % NTOT;
  out[2 * t + 0] = (float)(n / NHW);
  out[2 * t + 1] = (float)(n % NHW);
}

// ---------------------------------------------------------------------------
// Kernel 3: KNN over grid positions via WMMA Gram tiles.
// One wave per 16 queries. Per candidate tile: D = A(16x32 f16) x B(32x16 f16)
// with coords in K-slots 0,1 -> q.c dot products, exact in f32 accumulation.
// score = |c|^2 - 2 q.c  (|q|^2 dropped: constant per row, preserves ranking).
// Lanes 0-15 keep a sorted top-9 (slot 0 = self), then write edges for all 16
// batches (edge_index is batch-invariant).
// ---------------------------------------------------------------------------
__global__ __launch_bounds__(32) void knn_kernel(float* __restrict__ edge_out) {
  __shared__ float tile[16][17];   // padded to dodge bank conflicts

  const int lane = threadIdx.x;          // 0..31
  const int nn   = lane & 15;            // column of D this lane owns
  const bool qlow = lane < 16;
  const int qm = blockIdx.x * 16 + nn;   // query handled by lanes 0-15

  // A fragment: 16-bit A 16x32 layout — lanes 0-15 element e<8 -> K=e.
  // Only K=0 (x) and K=1 (y) nonzero; lanes 16-31 (K=8..15,24..31) all zero.
  int qi = qm / NHW, qj = qm % NHW;      // benign for padded qm >= NTOT
  v16h a = {};
  if (qlow) { a[0] = (_Float16)(float)qi; a[1] = (_Float16)(float)qj; }

  // top-9 (score, index) sorted ascending; ties keep smaller index (stable).
  float bd[KP1];
  int   bi[KP1];
#pragma unroll
  for (int s = 0; s < KP1; ++s) { bd[s] = 3.0e38f; bi[s] = 0x7fffffff; }

  for (int t = 0; t < QTILES; ++t) {
    const int cand = t * 16 + nn;
    float cx, cy;
    if (cand < NTOT) { cx = (float)(cand / NHW); cy = (float)(cand % NHW); }
    else             { cx = 20000.0f; cy = 20000.0f; }  // pushed out of top-k

    // B fragment: same per-lane K packing as A (lanes 0-15 element e -> K=e).
    v16h bmat = {};
    if (qlow) { bmat[0] = (_Float16)cx; bmat[1] = (_Float16)cy; }

    v8f acc = {};
    acc = __builtin_amdgcn_wmma_f32_16x16x32_f16(
        /*neg_a=*/false, a, /*neg_b=*/false, bmat,
        /*c_mod=*/(short)0, acc, /*reuse_a=*/false, /*reuse_b=*/false);

    const float cn2 = cx * cx + cy * cy;
    // D layout: lane L, VGPR r -> M = r + 8*(L>=16), N = L%16
    const int mbase = (lane >> 4) * 8;
#pragma unroll
    for (int r = 0; r < 8; ++r)
      tile[mbase + r][nn] = cn2 - 2.0f * acc[r];

    __syncthreads();

    if (qlow) {
      for (int n2 = 0; n2 < 16; ++n2) {
        const int ci = t * 16 + n2;
        if (ci >= NTOT) break;                  // uniform across the wave
        const float d = tile[nn][n2];
        if (d < bd[KP1 - 1]) {                  // fast-path reject
          bd[KP1 - 1] = d; bi[KP1 - 1] = ci;
#pragma unroll
          for (int s = KP1 - 1; s > 0; --s) {
            if (bd[s] < bd[s - 1]) {            // strict: ties stay stable
              float td = bd[s]; bd[s] = bd[s - 1]; bd[s - 1] = td;
              int   ti = bi[s]; bi[s] = bi[s - 1]; bi[s - 1] = ti;
            }
          }
        }
      }
    }
    __syncthreads();
  }

  // Emit edges: edge_index[b][0][q*8+j] = q, [b][1][q*8+j] = neighbor j+1.
  if (qlow && qm < NTOT) {
    const float src = (float)qm;
#pragma unroll
    for (int j = 0; j < KNN; ++j) {
      const float tgt = (float)bi[j + 1];       // slot 0 is self
      for (int b = 0; b < B_; ++b) {
        const size_t base = (size_t)b * (2u * NTOT * KNN);
        edge_out[base + (size_t)qm * KNN + j]                       = src;
        edge_out[base + (size_t)NTOT * KNN + (size_t)qm * KNN + j]  = tgt;
      }
    }
  }
}

// ---------------------------------------------------------------------------
extern "C" void kernel_launch(void* const* d_in, const int* in_sizes, int n_in,
                              void* d_out, int out_size, void* d_ws, size_t ws_size,
                              hipStream_t stream) {
  const float* x = (const float*)d_in[0];
  float* out = (float*)d_out;

  const size_t patches_elems = (size_t)B_ * NTOT * FEAT;   // 48,771,072
  const size_t pos_elems     = (size_t)B_ * NTOT * 2;      //    127,008
  float* pos_out  = out + patches_elems;
  float* edge_out = out + patches_elems + pos_elems;

  // im2col: 16*3969*3*16 groups * 4 float4s = 12,192,768 threads exactly
  patches_kernel<<<47628, 256, 0, stream>>>(x, out);

  pos_kernel<<<(B_ * NTOT + 255) / 256, 256, 0, stream>>>(pos_out);

  knn_kernel<<<QTILES, 32, 0, stream>>>(edge_out);
}
